// W4A16B16O16Linear_83416854823023
// MI455X (gfx1250) — matrix operand order repaired and checked by reading, not compiled, not run on hardware
//
#include <hip/hip_runtime.h>

// ---------------------------------------------------------------------------
// W4A16 GEMM for MI455X (gfx1250, wave32, WMMA 16x16x32 f16 -> f32 acc)
//   y[M,N] = x[M,K](f16) * dequant(q[K,N] int4-in-int32, scale[K/128,N] f16)
// Fixed problem: M = N = K = 4096, GROUP = 128, zero-point = 8.
// Double-buffered LDS pipeline, one barrier per K-tile; A tile staged via
// gfx1250 async global->LDS copies (GLOBAL_LOAD_ASYNC_TO_LDS_B128).
// ---------------------------------------------------------------------------

typedef __attribute__((ext_vector_type(16))) _Float16 v16h;
typedef __attribute__((ext_vector_type(8)))  _Float16 h8;
typedef __attribute__((ext_vector_type(2)))  _Float16 h2;
typedef __attribute__((ext_vector_type(8)))  float    v8f;
typedef __attribute__((ext_vector_type(4)))  int      v4i;

#define MDIM 4096
#define NDIM 4096
#define KDIM 4096
#define BM 128
#define BN 128
#define BK 64
#define LDA 72   // LDS stride (halves) for A tile, +8 pad vs 64 banks
#define LDB 72   // LDS stride (halves) for B^T tile

#if __has_builtin(__builtin_amdgcn_global_load_async_to_lds_b128)
#define USE_ASYNC_A 1
typedef __attribute__((address_space(1))) v4i* gas_v4i;
typedef __attribute__((address_space(3))) v4i* las_v4i;
#else
#define USE_ASYNC_A 0
#endif

__device__ __forceinline__ void wait_async_zero() {
#if USE_ASYNC_A
#if __has_builtin(__builtin_amdgcn_s_wait_asynccnt)
    __builtin_amdgcn_s_wait_asynccnt(0);
#else
    asm volatile("s_wait_asynccnt 0x0" ::: "memory");
#endif
#endif
}

union AFrag { v16h v; h8 h[2]; };
union WPack { h2 p[16]; h8 o[4]; };

__global__ __launch_bounds__(256, 2)
void w4a16_gemm_kernel(const _Float16* __restrict__ xp,
                       const _Float16* __restrict__ scalep,
                       const int*      __restrict__ qwp,
                       _Float16*       __restrict__ outp)
{
    __shared__ _Float16 Alds[2][BM * LDA];   // ping-pong x tiles
    __shared__ _Float16 Blds[2][BN * LDB];   // ping-pong dequantized W^T tiles

    const int tid  = threadIdx.x;
    const int bn   = blockIdx.x;
    const int bm   = blockIdx.y;

    const int lane = tid & 31;
    const int ll   = lane & 15;
    const int hi   = lane >> 4;
    const int wid  = tid >> 5;
    const int wm   = wid >> 2;            // 0..1  -> 64-row slab
    const int wn   = wid & 3;             // 0..3  -> 32-col slab

    // A: 2 threads per row, each covers 32 halves (4 x b128)
    const int arow = tid >> 1;
    const int acb  = (tid & 1) << 5;
    const _Float16* abase = xp + (size_t)(bm * BM + arow) * KDIM + acb;

    // B: one column of 32 consecutive K per thread (coalesced along N)
    const int ncol = tid & 127;
    const int kb   = (tid >> 7) << 5;     // 0 or 32
    const int gcol = bn * BN + ncol;
    const int* qbase = qwp + (size_t)kb * NDIM + gcol;

    v8f acc[4][2];
#pragma unroll
    for (int mt = 0; mt < 4; ++mt)
#pragma unroll
        for (int nt = 0; nt < 2; ++nt)
            acc[mt][nt] = (v8f)(0.0f);

    // -------- per-tile staging state (registers) ---------------------------
    int      qv[32];
#if !USE_ASYNC_A
    h8       areg[4];
#endif
    _Float16 sreg;

    // Issue global loads for K-tile k0; A goes straight to LDS buf nb when
    // the async path exists, otherwise into registers.
    auto issue_loads = [&](int k0, int nb) {
        const _Float16* asrc = abase + k0;
#if USE_ASYNC_A
#pragma unroll
        for (int j = 0; j < 4; ++j)
            __builtin_amdgcn_global_load_async_to_lds_b128(
                (gas_v4i)(v4i*)(asrc + j * 8),
                (las_v4i)(&Alds[nb][arow * LDA + acb + j * 8]), 0, 0);
#else
        (void)nb;
#pragma unroll
        for (int j = 0; j < 4; ++j)
            areg[j] = *(const h8*)(asrc + j * 8);
#endif
        const int* qsrc = qbase + (size_t)k0 * NDIM;
#pragma unroll
        for (int i = 0; i < 32; ++i)
            qv[i] = qsrc[(size_t)i * NDIM];
        sreg = scalep[(size_t)(k0 >> 7) * NDIM + gcol];
    };

    // Dequantize q regs and publish tile into LDS buffer nb.
    auto stage_tile = [&](int nb) {
#if !USE_ASYNC_A
#pragma unroll
        for (int j = 0; j < 4; ++j)
            *(h8*)(&Alds[nb][arow * LDA + acb + j * 8]) = areg[j];
#endif
        const _Float16 s = sreg;
        const _Float16 c = s * (_Float16)(-1032.0f);   // (q|0x6400)=1024+q ; -(1024+8)*s
        const h2 sv = {s, s};
        const h2 cv = {c, c};
        WPack w;
#pragma unroll
        for (int i = 0; i < 16; ++i) {
            unsigned p = (unsigned)qv[2 * i] | ((unsigned)qv[2 * i + 1] << 16)
                       | 0x64006400u;
            w.p[i] = __builtin_bit_cast(h2, p) * sv + cv;   // v_pk_fma_f16
        }
#pragma unroll
        for (int j = 0; j < 4; ++j)
            *(h8*)(&Blds[nb][ncol * LDB + kb + j * 8]) = w.o[j];
    };

    // 16 WMMAs on LDS buffer b.
    auto compute_tile = [&](int b) {
        const _Float16* Ab = &Alds[b][0];
        const _Float16* Bb = &Blds[b][0];
#pragma unroll
        for (int ks = 0; ks < BK; ks += 32) {
            AFrag a[4];
#pragma unroll
            for (int mt = 0; mt < 4; ++mt) {
                const _Float16* ap = Ab + (wm * 64 + mt * 16 + ll) * LDA + ks + hi * 8;
                a[mt].h[0] = *(const h8*)(ap);
                a[mt].h[1] = *(const h8*)(ap + 16);
            }
            AFrag bfr[2];
#pragma unroll
            for (int nt = 0; nt < 2; ++nt) {
                const _Float16* bp = Bb + (wn * 32 + nt * 16 + ll) * LDB + ks + hi * 8;
                bfr[nt].h[0] = *(const h8*)(bp);
                bfr[nt].h[1] = *(const h8*)(bp + 16);
            }
#pragma unroll
            for (int mt = 0; mt < 4; ++mt)
#pragma unroll
                for (int nt = 0; nt < 2; ++nt)
                    acc[mt][nt] = __builtin_amdgcn_wmma_f32_16x16x32_f16(
                        false, a[mt].v, false, bfr[nt].v,
                        (short)0, acc[mt][nt], false, false);
        }
    };

    // -------- prologue: stage tile 0 into buffer 0 -------------------------
    issue_loads(0, 0);
    stage_tile(0);
    wait_async_zero();
    __syncthreads();

    // -------- pipelined main loop: ONE barrier per K-tile ------------------
    int buf = 0;
    for (int k0 = 0; k0 < KDIM; k0 += BK) {
        const bool have_next = (k0 + BK) < KDIM;
        if (have_next)
            issue_loads(k0 + BK, buf ^ 1);   // async A -> other LDS buffer
        compute_tile(buf);                   // WMMA hides next-tile latency
        if (have_next) {
            stage_tile(buf ^ 1);             // dequant + publish next tile
            wait_async_zero();               // A async copies complete
        }
        __syncthreads();
        buf ^= 1;
    }

    // -------- epilogue: f32 acc -> f16 out ---------------------------------
#pragma unroll
    for (int mt = 0; mt < 4; ++mt) {
#pragma unroll
        for (int nt = 0; nt < 2; ++nt) {
            const int col = bn * BN + wn * 32 + nt * 16 + ll;
#pragma unroll
            for (int r = 0; r < 8; ++r) {
                const int row = bm * BM + wm * 64 + mt * 16 + hi * 8 + r;
                outp[(size_t)row * NDIM + col] = (_Float16)acc[mt][nt][r];
            }
        }
    }
}

extern "C" void kernel_launch(void* const* d_in, const int* in_sizes, int n_in,
                              void* d_out, int out_size, void* d_ws, size_t ws_size,
                              hipStream_t stream)
{
    (void)in_sizes; (void)n_in; (void)d_ws; (void)ws_size; (void)out_size;
    const _Float16* x     = (const _Float16*)d_in[0];
    const _Float16* scale = (const _Float16*)d_in[1];
    const int*      qw    = (const int*)d_in[2];
    _Float16*       out   = (_Float16*)d_out;

    dim3 grid(NDIM / BN, MDIM / BM);   // 32 x 32 blocks
    dim3 block(256);                   // 8 wave32
    w4a16_gemm_kernel<<<grid, block, 0, stream>>>(x, scale, qw, out);
}
